// LifelongGenerator_35330400977491
// MI455X (gfx1250) — matrix operand order repaired and checked by reading, not compiled, not run on hardware
//
#include <hip/hip_runtime.h>
#include <math.h>

typedef __attribute__((ext_vector_type(16))) _Float16 v16h;
typedef __attribute__((ext_vector_type(8)))  _Float16 v8h;
typedef __attribute__((ext_vector_type(8)))  float    v8f;
typedef int v4i __attribute__((__vector_size__(16)));   // matches async builtin param

#define SQRT2F 1.41421356237309515f

// Async load-to-LDS path (CDNA5 ASYNCcnt), with safe fallback if the
// toolchain doesn't declare the builtins.
#if defined(__AMDGCN__) && __has_builtin(__builtin_amdgcn_global_load_async_to_lds_b128) && __has_builtin(__builtin_amdgcn_s_wait_asynccnt)
#define USE_ASYNC_LDS 1
#else
#define USE_ASYNC_LDS 0
#endif

// ------------------------------------------------------------------
// Style MLP: pixelnorm + 8 modulated EqualLinear layers (lr_mul=0.01)
// ------------------------------------------------------------------
struct MlpParams {
  const float* w[8];   const float* b[8];
  const float* wmi[8]; const float* wmo[8];
  const float* wai[8]; const float* wao[8];
  const float* bm[8];  const float* ba[8];
};

__global__ __launch_bounds__(512)
void mlp_kernel(const float* __restrict__ z, MlpParams P, float* __restrict__ lat_out) {
  __shared__ float lat[2][512];
  __shared__ float red[512];
  __shared__ float norm[2];
  int t = threadIdx.x;
  for (int b = 0; b < 2; ++b) {
    float v = z[b * 512 + t];
    red[t] = v * v;
    __syncthreads();
    for (int s2 = 256; s2 > 0; s2 >>= 1) {
      if (t < s2) red[t] += red[t + s2];
      __syncthreads();
    }
    if (t == 0) norm[b] = rsqrtf(red[0] * (1.0f / 512.0f) + 1e-8f);
    __syncthreads();
    lat[b][t] = v * norm[b];
    __syncthreads();
  }
  const float scale = 0.01f * 0.04419417382415922f; // lr_mul / sqrt(512)
  for (int L = 0; L < 8; ++L) {
    const float* w   = P.w[L];
    const float* wmi = P.wmi[L];
    const float* wai = P.wai[L];
    float mo[4], ao[4];
#pragma unroll
    for (int r = 0; r < 4; ++r) { mo[r] = P.wmo[L][t*4+r]; ao[r] = P.wao[L][t*4+r]; }
    float a0 = 0.f, a1 = 0.f;
    for (int j = 0; j < 512; ++j) {
      float m = 0.f, ad = 0.f;
#pragma unroll
      for (int r = 0; r < 4; ++r) { m += mo[r]*wmi[j*4+r]; ad += ao[r]*wai[j*4+r]; }
      float wh = (w[t*512 + j] * m + ad) * scale;
      a0 += lat[0][j] * wh;
      a1 += lat[1][j] * wh;
    }
    float bh = (P.b[L][t] * P.bm[L][t] + P.ba[L][t]) * 0.01f;
    float y0 = a0 + bh, y1 = a1 + bh;
    y0 = (y0 >= 0.f ? y0 : 0.2f * y0) * SQRT2F;
    y1 = (y1 >= 0.f ? y1 : 0.2f * y1) * SQRT2F;
    __syncthreads();
    lat[0][t] = y0; lat[1][t] = y1;
    __syncthreads();
  }
  lat_out[t] = lat[0][t];
  lat_out[512 + t] = lat[1][t];
}

// ------------------------------------------------------------------
// Style projection s = EqualLinear(lat) for one conv's smod (lr_mul=1)
// ------------------------------------------------------------------
__global__ __launch_bounds__(256)
void style_kernel(const float* __restrict__ lat,
                  const float* __restrict__ w,  const float* __restrict__ bias,
                  const float* __restrict__ wmi, const float* __restrict__ wmo,
                  const float* __restrict__ wai, const float* __restrict__ wao,
                  const float* __restrict__ bm,  const float* __restrict__ ba,
                  float* __restrict__ s, int cin) {
  int i = blockIdx.x * 256 + threadIdx.x;
  if (i >= cin) return;
  float mo[4], ao[4];
#pragma unroll
  for (int r = 0; r < 4; ++r) { mo[r] = wmo[i*4+r]; ao[r] = wao[i*4+r]; }
  const float scale = 0.04419417382415922f; // 1/sqrt(512)
  float a0 = 0.f, a1 = 0.f;
  for (int j = 0; j < 512; ++j) {
    float m = 0.f, ad = 0.f;
#pragma unroll
    for (int r = 0; r < 4; ++r) { m += mo[r]*wmi[j*4+r]; ad += ao[r]*wai[j*4+r]; }
    float wh = (w[i*512 + j] * m + ad) * scale;
    a0 += lat[j] * wh;
    a1 += lat[512 + j] * wh;
  }
  float bh = bias[i] * bm[i] + ba[i];
  s[i]       = a0 + bh;
  s[cin + i] = a1 + bh;
}

// ------------------------------------------------------------------
// Conv weight modulation: rank-4 LeFT conv modulator + style scale + demod.
// Output stored as f16 (WMMA operand precision) to halve weight bandwidth.
// ------------------------------------------------------------------
__global__ __launch_bounds__(256)
void modw_kernel(const float* __restrict__ w,
                 const float* __restrict__ m1o, const float* __restrict__ m1i,
                 const float* __restrict__ a1o, const float* __restrict__ a1i,
                 const float* __restrict__ m2i, const float* __restrict__ a2i,
                 const float* __restrict__ a2inst,
                 const float* __restrict__ s, _Float16* __restrict__ wb,
                 int cin, int cout, int KK, int demod) {
  int o = blockIdx.x, b = blockIdx.y, t = threadIdx.x;
  __shared__ float rm1[4][9];
  __shared__ float red[256];
  if (t < 4 * KK) {
    int rb = t / KK, kk = t % KK;
    float a1 = 0.f, m = 0.f;
#pragma unroll
    for (int r = 0; r < 4; ++r) {
      a1 += a1o[o*4 + r] * a1i[kk*4 + r];
      m  += m1o[o*4 + r] * m1i[(rb*4 + r) * KK + kk];
    }
    float v = m + a1;
    rm1[rb][kk] = v > 0.f ? v : 0.f;
  }
  __syncthreads();
  int K = cin * KK;
  float scale = rsqrtf((float)K);
  float vreg[18];
  float acc = 0.f;
  int cnt = 0;
  for (int e = t; e < K; e += 256) {
    int i = e / KK, kk = e % KK;
    float m2 = 0.f, a2 = 0.f;
#pragma unroll
    for (int r = 0; r < 4; ++r) {
      m2 += m2i[i*4 + r] * rm1[r][kk];
      a2 += a2i[i*4 + r] * a2inst[kk*4 + r];
    }
    float wh = w[(size_t)o * K + e] * m2 + a2;
    float v = wh * s[b * cin + i] * scale;
    vreg[cnt++] = v;
    acc += v * v;
  }
  red[t] = acc;
  __syncthreads();
  for (int st = 128; st > 0; st >>= 1) {
    if (t < st) red[t] += red[t + st];
    __syncthreads();
  }
  float d = demod ? rsqrtf(red[0] + 1e-8f) : 1.0f;
  cnt = 0;
  for (int e = t; e < K; e += 256)
    wb[((size_t)b * cout + o) * K + e] = (_Float16)(vreg[cnt++] * d);
}

// ------------------------------------------------------------------
// 3x3 conv (mode 0: pad=1) / transpose conv as dilated gather (mode 1).
// Implicit GEMM via v_wmma_f32_16x16x32_f16:
//   block = 128 threads (4 waves), M = 32 out channels (2 A fragments),
//   each wave owns a 16-pixel N tile sharing one B fragment -> 2 WMMA/iter.
//   Double-buffered LDS; f16 weight tiles moved global->LDS with
//   GLOBAL_LOAD_ASYNC_TO_LDS_B128 (ASYNCcnt) when available;
//   branch-free clamped im2col gathers.
// ------------------------------------------------------------------
__global__ __launch_bounds__(128)
void conv3_wmma_kernel(const float* __restrict__ xin, const _Float16* __restrict__ wb,
                       float* __restrict__ out,
                       int cin, int cout, int H, int W, int OH, int OW, int mode) {
  __shared__ __attribute__((aligned(16))) _Float16 wT[2][32][32];   // A tiles (32M x 32K)
  __shared__ __attribute__((aligned(32))) _Float16 xT[2][4][32][16];// B tiles per wave
  const int tid  = threadIdx.x;
  const int wave = tid >> 5, lane = tid & 31;
  const int b = blockIdx.z;
  const int oBase = blockIdx.y * 32;
  const int pBase = (blockIdx.x * 4 + wave) * 16;
  const int NP = OH * OW;
  const int Ktot = cin * 9;
  const int nK = Ktot >> 5;
  const _Float16* wrow = wb + (size_t)b * cout * Ktot + (size_t)oBase * Ktot;
  const float* xb = xin + (size_t)b * cin * H * W;

  // pixel coordinates for this wave's 16 pixels: loop-invariant, one div total
  int py[16], px[16];
  {
    int y0 = pBase / OW, x0 = pBase - y0 * OW;
#pragma unroll
    for (int n = 0; n < 16; ++n) {
      if (pBase + n < NP) { py[n] = y0; px[n] = x0; }
      else                { py[n] = -0x4000; px[n] = -0x4000; }
      if (++x0 == OW) { x0 = 0; ++y0; }
    }
  }

  // incremental K decomposition for this lane's gather row (k advances by 32)
  int ic = lane / 9;        // input channel
  int rr = lane - ic * 9;   // 3x3 tap index 0..8

  // per-thread weight-tile slice: one 16-byte (8 x f16) chunk
  const int wrowi = tid >> 2;            // 0..31
  const int wcolb = (tid & 3) * 8;       // 0,8,16,24

  auto stage = [&](int buf, int k0) {
    // ---- weight tile: 32x32 f16, one b128 move per thread ----
    const _Float16* gsrc = &wrow[(size_t)wrowi * Ktot + k0 + wcolb];
#if USE_ASYNC_LDS
    __builtin_amdgcn_global_load_async_to_lds_b128(
        (v4i*)gsrc, (v4i*)&wT[buf][wrowi][wcolb], 0, 0);
#else
    *(v8h*)&wT[buf][wrowi][wcolb] = *(const v8h*)gsrc;
#endif
    // prefetch the tile after next into cache (global_prefetch_b8)
    if (k0 + 64 < Ktot)
      __builtin_prefetch(&wrow[(size_t)wrowi * Ktot + k0 + 64], 0, 1);
    // ---- im2col row: lane = one (channel, tap) K row; branch-free ----
    int ky = (rr * 11) >> 5;            // rr / 3
    int kx = rr - ky * 3;               // rr % 3
    const float* xi = xb + (size_t)ic * H * W;
#pragma unroll
    for (int n = 0; n < 16; ++n) {
      int u, vx;
      bool ok;
      if (mode == 0) {
        u = py[n] + ky - 1; vx = px[n] + kx - 1;
        ok = (u >= 0) & (u < H) & (vx >= 0) & (vx < W);
      } else {
        int uu = py[n] - ky, vv = px[n] - kx;
        ok = ((uu | vv) >= 0) & (((uu | vv) & 1) == 0);
        u = uu >> 1; vx = vv >> 1;
        ok = ok & (u < H) & (vx < W);
      }
      int ui = ok ? u : 0;
      int vi = ok ? vx : 0;
      float v = xi[ui * W + vi];          // unconditional, clamped address
      v = ok ? v : 0.f;
      (&xT[buf][wave][0][0])[n * 32 + lane] = (_Float16)v;
    }
    // advance (channel, tap) by 32 K steps: 32 = 3*9 + 5
    ic += 3; rr += 5;
    if (rr >= 9) { rr -= 9; ic += 1; }
  };

  v8f c0 = {}, c1 = {};
  stage(0, 0);
#if USE_ASYNC_LDS
  __builtin_amdgcn_s_wait_asynccnt(0);
#endif
  __syncthreads();
  for (int kt = 0; kt < nK; ++kt) {
    int cb = kt & 1;
    if (kt + 1 < nK) stage(cb ^ 1, (kt + 1) << 5);
    // A fragments per CDNA5 16-bit A(16x32) layout; B: lane = N, K in same order as A
    int m = lane & 15, half = lane >> 4;
    v8h lo0 = *(const v8h*)&wT[cb][m][8 * half];
    v8h hi0 = *(const v8h*)&wT[cb][m][16 + 8 * half];
    v8h lo1 = *(const v8h*)&wT[cb][16 + m][8 * half];
    v8h hi1 = *(const v8h*)&wT[cb][16 + m][16 + 8 * half];
    v16h a0 = __builtin_shufflevector(lo0, hi0, 0,1,2,3,4,5,6,7,8,9,10,11,12,13,14,15);
    v16h a1 = __builtin_shufflevector(lo1, hi1, 0,1,2,3,4,5,6,7,8,9,10,11,12,13,14,15);
    const _Float16* xs = &xT[cb][wave][0][0] + m * 32;
    v8h blo = *(const v8h*)&xs[8 * half];
    v8h bhi = *(const v8h*)&xs[16 + 8 * half];
    v16h bf = __builtin_shufflevector(blo, bhi, 0,1,2,3,4,5,6,7,8,9,10,11,12,13,14,15);
    c0 = __builtin_amdgcn_wmma_f32_16x16x32_f16(false, a0, false, bf, (short)0, c0, false, false);
    c1 = __builtin_amdgcn_wmma_f32_16x16x32_f16(false, a1, false, bf, (short)0, c1, false, false);
#if USE_ASYNC_LDS
    __builtin_amdgcn_s_wait_asynccnt(0);   // drain async weight copy for buf^1
#endif
    __syncthreads();
  }
  // D layout: lane holds N = lane&15; VGPR v -> M = v + 8*(lane>>4)
  int n = lane & 15, rh = lane >> 4;
  int p = pBase + n;
  if (p < NP) {
    float* ob = out + (size_t)b * cout * NP;
#pragma unroll
    for (int v = 0; v < 8; ++v) {
      ob[(size_t)(oBase + rh * 8 + v) * NP + p]      = c0[v];
      ob[(size_t)(oBase + 16 + rh * 8 + v) * NP + p] = c1[v];
    }
  }
}

// ------------------------------------------------------------------
// Blur after transpose conv: 4x4 [1,3,3,1] outer kernel *4, pad (1,1)
// ------------------------------------------------------------------
__global__ void blur_kernel(const float* __restrict__ in, float* __restrict__ out,
                            int C, int S, int OS) {
  int idx = blockIdx.x * blockDim.x + threadIdx.x;
  int total = 2 * C * OS * OS;
  if (idx >= total) return;
  int x = idx % OS, y = (idx / OS) % OS, bc = idx / (OS * OS);
  const float* p = in + (size_t)bc * S * S;
  const float k1[4] = {1.f, 3.f, 3.f, 1.f};
  float acc = 0.f;
#pragma unroll
  for (int dy = 0; dy < 4; ++dy) {
    int u = y - 1 + dy;
    if (u < 0 || u >= S) continue;
#pragma unroll
    for (int dx = 0; dx < 4; ++dx) {
      int v = x - 1 + dx;
      if (v < 0 || v >= S) continue;
      acc += k1[dy] * k1[dx] * p[u * S + v];
    }
  }
  out[idx] = acc * (1.f / 16.f);
}

// ------------------------------------------------------------------
// x = fused_leaky_relu(x + noise_w*noise, act_bias)   (in place)
// ------------------------------------------------------------------
__global__ void noise_act_kernel(float* __restrict__ x, const float* __restrict__ noise,
                                 const float* __restrict__ nw,
                                 const float* __restrict__ ab, int C, int HW) {
  int idx = blockIdx.x * blockDim.x + threadIdx.x;
  int total = 2 * C * HW;
  if (idx >= total) return;
  int p = idx % HW, c = (idx / HW) % C;
  float v = x[idx] + nw[0] * noise[p] + ab[c];
  x[idx] = (v >= 0.f ? v : 0.2f * v) * SQRT2F;
}

// ------------------------------------------------------------------
// toRGB: 1x1 conv (no demod, f16 weights) + bias + upsampled prev skip
// ------------------------------------------------------------------
__global__ void torgb_kernel(const float* __restrict__ x, const _Float16* __restrict__ wb,
                             const float* __restrict__ bias,
                             const float* __restrict__ skipPrev,
                             float* __restrict__ skipOut, int cin, int H, int W) {
  int idx = blockIdx.x * blockDim.x + threadIdx.x;
  int total = 2 * 3 * H * W;
  if (idx >= total) return;
  int p = idx % (H * W);
  int c = (idx / (H * W)) % 3;
  int b = idx / (3 * H * W);
  int y = p / W, xx = p % W;
  const float* xb = x + (size_t)b * cin * H * W + p;
  const _Float16* wr = wb + ((size_t)b * 3 + c) * cin;
  float acc = 0.f;
  for (int i = 0; i < cin; ++i) acc += (float)wr[i] * xb[(size_t)i * H * W];
  acc += bias[c];
  if (skipPrev) {
    int Hh = H >> 1, Wh = W >> 1;
    const float* sp = skipPrev + ((size_t)b * 3 + c) * Hh * Wh;
    const float k1[4] = {1.f, 3.f, 3.f, 1.f};
    float up = 0.f;
#pragma unroll
    for (int dy = 0; dy < 4; ++dy) {
      int u = y + dy - 2;
      if (u < 0 || (u & 1)) continue;
      u >>= 1;
      if (u >= Hh) continue;
#pragma unroll
      for (int dx = 0; dx < 4; ++dx) {
        int v = xx + dx - 2;
        if (v < 0 || (v & 1)) continue;
        v >>= 1;
        if (v >= Wh) continue;
        up += k1[dy] * k1[dx] * sp[u * Wh + v];
      }
    }
    acc += up * (1.f / 16.f);
  }
  skipOut[idx] = acc;
}

__global__ void tile_kernel(const float* __restrict__ src, float* __restrict__ dst, int n) {
  int i = blockIdx.x * blockDim.x + threadIdx.x;
  if (i < 2 * n) dst[i] = src[i % n];
}

// ------------------------------------------------------------------
// Host orchestration
// ------------------------------------------------------------------
struct ModConv {
  const float *w;
  const float *sw, *sb, *smi, *smo, *sai, *sao, *sbm, *sba;   // smod (eql)
  const float *m1o, *m1i, *a1o, *a1i, *m2i, *a2i, *a2inst;    // wmod
};

extern "C" void kernel_launch(void* const* d_in, const int* in_sizes, int n_in,
                              void* d_out, int out_size, void* d_ws, size_t ws_size,
                              hipStream_t stream) {
  (void)in_sizes; (void)n_in; (void)out_size; (void)ws_size;
  auto F = [&](int i) -> const float* { return (const float*)d_in[i]; };
  auto MC = [&](int base) {
    ModConv m;
    m.w   = F(base + 0);
    m.sw  = F(base + 1);  m.sb  = F(base + 2);
    m.smi = F(base + 3);  m.smo = F(base + 4);
    m.sai = F(base + 5);  m.sao = F(base + 6);
    m.sbm = F(base + 7);  m.sba = F(base + 8);
    m.m1o = F(base + 9);  m.m1i = F(base + 10);
    m.a1o = F(base + 11); m.a1i = F(base + 12);
    m.m2i = F(base + 13); m.a2i = F(base + 14); m.a2inst = F(base + 15);
    return m;
  };

  // workspace layout (floats)
  float* WS    = (float*)d_ws;
  float* lat   = WS;                    // 1024
  float* sbuf  = WS + 1024;             // 1024
  _Float16* wbb = (_Float16*)(WS + 2048); // 2*512*4608 f16 (fits in 2359296 floats)
  float* actA  = WS + 2048 + 4718592;   // 8520192 (fits 2x256x129x129)
  float* actB  = actA + 8520192;        // 8520192
  float* skipA = actB + 8520192;        // 98304
  float* skipB = skipA + 98304;         // 98304

  // 1) style MLP
  MlpParams mp;
  for (int L = 0; L < 8; ++L) {
    int base = 1 + 8 * L;
    mp.w[L]   = F(base + 0); mp.b[L]   = F(base + 1);
    mp.wmi[L] = F(base + 2); mp.wmo[L] = F(base + 3);
    mp.wai[L] = F(base + 4); mp.wao[L] = F(base + 5);
    mp.bm[L]  = F(base + 6); mp.ba[L]  = F(base + 7);
  }
  mlp_kernel<<<1, 512, 0, stream>>>(F(0), mp, lat);

  // 2) tile const (1,512,4,4) -> (2,512,4,4)
  tile_kernel<<<(16384 + 255) / 256, 256, 0, stream>>>(F(65), actA, 8192);

  auto style_modw = [&](const ModConv& m, int cin, int cout, int KK, int demod) {
    style_kernel<<<dim3((cin + 255) / 256), 256, 0, stream>>>(
        lat, m.sw, m.sb, m.smi, m.smo, m.sai, m.sao, m.sbm, m.sba, sbuf, cin);
    modw_kernel<<<dim3(cout, 2), 256, 0, stream>>>(
        m.w, m.m1o, m.m1i, m.a1o, m.a1i, m.m2i, m.a2i, m.a2inst,
        sbuf, wbb, cin, cout, KK, demod);
  };

  // 3) conv1 (styled, 512->512, 4x4, mode 0) : indices 66..83
  {
    ModConv c1 = MC(66);
    style_modw(c1, 512, 512, 9, 1);
    conv3_wmma_kernel<<<dim3(1, 512 / 32, 2), 128, 0, stream>>>(
        actA, wbb, actB, 512, 512, 4, 4, 4, 4, 0);
    int tot = 2 * 512 * 16;
    noise_act_kernel<<<(tot + 255) / 256, 256, 0, stream>>>(
        actB, F(366), F(82), F(83), 512, 16);
  }

  // 4) to_rgb1 : indices 84..100
  {
    ModConv r1 = MC(84);
    style_modw(r1, 512, 3, 1, 0);
    int tot = 2 * 3 * 16;
    torgb_kernel<<<(tot + 255) / 256, 256, 0, stream>>>(
        actB, wbb, F(100), nullptr, skipA, 512, 4, 4);
  }

  // 5) levels
  float* cur = actB;
  float* tmp = actA;
  float* skcur = skipA;
  const int cins[5]  = {512, 512, 512, 512, 512};
  const int couts[5] = {512, 512, 512, 512, 256};
  for (int l = 0; l < 5; ++l) {
    int base = 101 + 53 * l;
    int res = 8 << l;
    int Hin = res >> 1;
    int cin = cins[l], cout = couts[l];
    const float* noiseUp = F(367 + 2 * l);
    const float* noiseCv = F(368 + 2 * l);

    // -- up styled (transpose conv + blur + noise/act)
    {
      ModConv mu = MC(base);
      style_modw(mu, cin, cout, 9, 1);
      int S = 2 * Hin + 1;
      int NPt = S * S;
      conv3_wmma_kernel<<<dim3((NPt + 63) / 64, cout / 32, 2), 128, 0, stream>>>(
          cur, wbb, tmp, cin, cout, Hin, Hin, S, S, 1);
      int tot = 2 * cout * res * res;
      blur_kernel<<<(tot + 255) / 256, 256, 0, stream>>>(tmp, cur, cout, S, res);
      noise_act_kernel<<<(tot + 255) / 256, 256, 0, stream>>>(
          cur, noiseUp, F(base + 16), F(base + 17), cout, res * res);
    }
    // -- regular styled conv
    {
      ModConv mc2 = MC(base + 18);
      style_modw(mc2, cout, cout, 9, 1);
      int NP = res * res;
      conv3_wmma_kernel<<<dim3((NP + 63) / 64, cout / 32, 2), 128, 0, stream>>>(
          cur, wbb, tmp, cout, cout, res, res, res, res, 0);
      int tot = 2 * cout * NP;
      noise_act_kernel<<<(tot + 255) / 256, 256, 0, stream>>>(
          tmp, noiseCv, F(base + 34), F(base + 35), cout, NP);
      float* t = cur; cur = tmp; tmp = t;
    }
    // -- toRGB with skip upsample
    {
      ModConv mr = MC(base + 36);
      style_modw(mr, cout, 3, 1, 0);
      float* dst = (l == 4) ? (float*)d_out : ((skcur == skipA) ? skipB : skipA);
      int tot = 2 * 3 * res * res;
      torgb_kernel<<<(tot + 255) / 256, 256, 0, stream>>>(
          cur, wbb, F(base + 52), skcur, dst, cout, res, res);
      skcur = dst;
    }
  }
}
